// Attention_79164837200433
// MI455X (gfx1250) — matrix operand order repaired and checked by reading, not compiled, not run on hardware
//
#include <hip/hip_runtime.h>
#include <hip/hip_bf16.h>

#define B_DIM 4
#define S_DIM 2048
#define D_DIM 1024

typedef __attribute__((ext_vector_type(16))) __bf16 v16bf;
typedef __attribute__((ext_vector_type(8)))  float  v8f;

union BFrag {
    v16bf v;
    unsigned u[8];
    uint4 q[2];
};

__device__ __forceinline__ unsigned short f2bf(float f) {
    unsigned u = __float_as_uint(f);
    unsigned r = (u + 0x7FFFu + ((u >> 16) & 1u)) >> 16;
    return (unsigned short)r;
}

// A-fragment (16-bit A, 16x32): lane m holds K = 2r + (r>=4?8:0) + 8g.
// As two 16B chunks: element offsets {8g .. 8g+7} and {16+8g .. 23+8g}.
__device__ __forceinline__ void loadA(BFrag& f, const unsigned short* p, int g) {
    f.q[0] = *(const uint4*)(p + 8 * g);
    f.q[1] = *(const uint4*)(p + 16 + 8 * g);
}
// B-fragment (16-bit B, 32x16): same K order as A.
// As two 16B chunks: element offsets {8g .. 8g+7} and {16+8g .. 23+8g}.
__device__ __forceinline__ void loadB(BFrag& f, const unsigned short* p, int g) {
    f.q[0] = *(const uint4*)(p + 8 * g);
    f.q[1] = *(const uint4*)(p + 16 + 8 * g);
}

// ---------------- Stage 1: f32 -> bf16 conversion ----------------
__global__ void cvt_f32_bf16(const float* __restrict__ src,
                             unsigned short* __restrict__ dst, int n) {
    int i = blockIdx.x * blockDim.x + threadIdx.x;
    if (i < n) dst[i] = f2bf(src[i]);
}

// ---------------- Stage 2: QKV projection (y = x @ W^T + b) ----------------
// One wave computes a 64x64 output block (4x4 WMMA tiles, 128 acc VGPRs).
// which: 0=Q, 1=K, 2=V (stored transposed into Vt[b][d][s]).
__global__ __launch_bounds__(256)
void qkv_proj(const unsigned short* __restrict__ xb,
              const unsigned short* __restrict__ Wqb,
              const unsigned short* __restrict__ Wkb,
              const unsigned short* __restrict__ Wvb,
              const float* __restrict__ bq,
              const float* __restrict__ bk,
              const float* __restrict__ bv,
              unsigned short* __restrict__ Qb,
              unsigned short* __restrict__ Kb,
              unsigned short* __restrict__ Vt) {
    const int lane = threadIdx.x & 31;
    const int wv   = threadIdx.x >> 5;
    const int gw   = blockIdx.x * 8 + wv;   // global wave id, 6144 total
    const int which = gw >> 11;             // 2048 waves per matrix
    const int t  = gw & 2047;
    const int rb = t >> 4;                  // row block: 0..127 (64 rows each)
    const int eb = t & 15;                  // e block: 0..15 (64 cols each)
    const int g  = lane >> 4;
    const int mn = lane & 15;
    const int m0 = rb * 64;
    const int e0 = eb * 64;

    const unsigned short* W = (which == 0) ? Wqb : ((which == 1) ? Wkb : Wvb);
    const float* bias       = (which == 0) ? bq  : ((which == 1) ? bk  : bv);

    v8f acc[16];
#pragma unroll
    for (int i = 0; i < 16; i++) acc[i] = v8f{};

    for (int kk = 0; kk < D_DIM; kk += 32) {
        BFrag a[4];
#pragma unroll
        for (int i = 0; i < 4; i++)
            loadA(a[i], xb + (size_t)(m0 + i * 16 + mn) * D_DIM + kk, g);
#pragma unroll
        for (int e = 0; e < 4; e++) {
            BFrag b;
            // B[k][n] = W[n,k] (W row-major (e,d)) -> contiguous in k
            loadB(b, W + (size_t)(e0 + e * 16 + mn) * D_DIM + kk, g);
#pragma unroll
            for (int i = 0; i < 4; i++)
                acc[i * 4 + e] = __builtin_amdgcn_wmma_f32_16x16x32_bf16(
                    false, a[i].v, false, b.v, (short)0, acc[i * 4 + e], false, false);
        }
    }

#pragma unroll
    for (int e = 0; e < 4; e++) {
        const float bval = bias[e0 + e * 16 + mn];
        if (which == 2) {
            const int bb = m0 / S_DIM;     // 64 | 2048, so block is in one batch
#pragma unroll
            for (int i = 0; i < 4; i++)
#pragma unroll
                for (int r = 0; r < 8; r++) {
                    int s = (m0 % S_DIM) + i * 16 + r + 8 * g;
                    Vt[((size_t)bb * D_DIM + e0 + e * 16 + mn) * S_DIM + s] =
                        f2bf(acc[i * 4 + e][r] + bval);
                }
        } else {
            unsigned short* O = (which == 0) ? Qb : Kb;
#pragma unroll
            for (int i = 0; i < 4; i++)
#pragma unroll
                for (int r = 0; r < 8; r++) {
                    int row = m0 + i * 16 + r + 8 * g;
                    O[(size_t)row * D_DIM + e0 + e * 16 + mn] =
                        f2bf(acc[i * 4 + e][r] + bval);
                }
        }
    }
}

// ---------------- Stage 3: flash attention ----------------
// One block (8 waves) per 16-query tile. Per 128-key chunk:
//  - wave w computes the FULL score tile for keys [j0+16w, j0+16w+16) over all D
//    (no cross-wave partial reduction), Q streamed from LDS.
//  - cross-wave sharing is only 16 rowmax/rowsum scalars + the 16x128 bf16 p tile.
//  - wave w owns D columns [w*128, (w+1)*128) of the output (8 acc tiles).
#define QS_STRIDE 1032   // 1024 + 8 pad; row = 2064 B (16B-mult, banks spread)
#define PS_STRIDE 136    // 128 + 8 pad;  row = 272 B  (16B-mult, banks spread)

__global__ __launch_bounds__(256)
void flash_attn(const unsigned short* __restrict__ Qb,
                const unsigned short* __restrict__ Kb,
                const unsigned short* __restrict__ Vt,
                float* __restrict__ out) {
    __shared__ __align__(16) unsigned short Qs[16 * QS_STRIDE];   // 33.0 KB
    __shared__ __align__(16) unsigned short pshare[16 * PS_STRIDE]; // 4.3 KB
    __shared__ float rowmaxL[8][16];
    __shared__ float rowsumL[8][16];

    const int lane = threadIdx.x & 31;
    const int wv   = threadIdx.x >> 5;
    const int g    = lane >> 4;
    const int mn   = lane & 15;

    const int qblk = blockIdx.x;          // 0..511
    const int bb   = qblk >> 7;
    const int q0   = (qblk & 127) * 16;

    const unsigned short* Qrow  = Qb + ((size_t)bb * S_DIM + q0) * D_DIM;
    const unsigned short* Kbase = Kb + (size_t)bb * S_DIM * D_DIM;
    const unsigned short* Vbase = Vt + (size_t)bb * D_DIM * S_DIM;

    const int d0 = wv * 128;              // this wave's output D-slice
    const float scale = 0.03125f;         // 1/sqrt(1024)

    // cooperative Q staging: 16 x 1024 bf16 -> LDS (padded stride)
    for (int idx = threadIdx.x; idx < 16 * 128; idx += 256) {
        int row = idx >> 7;
        int c8  = idx & 127;
        uint4 v = *(const uint4*)(Qrow + (size_t)row * D_DIM + c8 * 8);
        *(uint4*)(Qs + row * QS_STRIDE + c8 * 8) = v;
    }

    v8f acc[8];
#pragma unroll
    for (int i = 0; i < 8; i++) acc[i] = v8f{};
    float mrow[8], lrow[8];
#pragma unroll
    for (int r = 0; r < 8; r++) { mrow[r] = -3.0e38f; lrow[r] = 0.0f; }

    __syncthreads();

    for (int j0 = 0; j0 < S_DIM; j0 += 128) {
        const int ky0 = j0 + wv * 16;     // this wave's 16 keys

        // ---- full score tile for my 16 keys over all of D (32 WMMAs) ----
        v8f s = {};
        for (int c = 0; c < D_DIM; c += 32) {
            BFrag qa, kf;
            loadA(qa, Qs + mn * QS_STRIDE + c, g);
            loadB(kf, Kbase + (size_t)(ky0 + mn) * D_DIM + c, g);
            s = __builtin_amdgcn_wmma_f32_16x16x32_bf16(
                false, qa.v, false, kf.v, (short)0, s, false, false);
        }

        // ---- per-tile row max (rows r+8g; reduce across 16-lane groups) ----
#pragma unroll
        for (int r = 0; r < 8; r++) {
            s[r] *= scale;
            float x = s[r];
            x = fmaxf(x, __shfl_xor(x, 1));
            x = fmaxf(x, __shfl_xor(x, 2));
            x = fmaxf(x, __shfl_xor(x, 4));
            x = fmaxf(x, __shfl_xor(x, 8));
            if (mn == 0) rowmaxL[wv][r + 8 * g] = x;
        }
        __syncthreads();

        // ---- combine row maxes across waves, exponentiate, share p ----
        float alpha[8];
#pragma unroll
        for (int r = 0; r < 8; r++) {
            const int row = r + 8 * g;
            float cm = rowmaxL[0][row];
#pragma unroll
            for (int w = 1; w < 8; w++) cm = fmaxf(cm, rowmaxL[w][row]);
            float mnew = fmaxf(mrow[r], cm);
            alpha[r] = __expf(mrow[r] - mnew);
            mrow[r]  = mnew;
            float p = __expf(s[r] - mnew);
            s[r] = p;
            float rs = p;
            rs += __shfl_xor(rs, 1);
            rs += __shfl_xor(rs, 2);
            rs += __shfl_xor(rs, 4);
            rs += __shfl_xor(rs, 8);
            if (mn == 0) rowsumL[wv][row] = rs;
            pshare[row * PS_STRIDE + wv * 16 + mn] = f2bf(p);
        }
        __syncthreads();

        // ---- update l, rescale acc ----
#pragma unroll
        for (int r = 0; r < 8; r++) {
            const int row = r + 8 * g;
            float cs = 0.f;
#pragma unroll
            for (int w = 0; w < 8; w++) cs += rowsumL[w][row];
            lrow[r] = lrow[r] * alpha[r] + cs;
        }
#pragma unroll
        for (int i = 0; i < 8; i++)
#pragma unroll
            for (int r = 0; r < 8; r++) acc[i][r] *= alpha[r];

        // ---- load p as 4 A-fragments (16x128 keys) ----
        BFrag pA[4];
#pragma unroll
        for (int f = 0; f < 4; f++)
            loadA(pA[f], pshare + mn * PS_STRIDE + f * 32, g);
        __syncthreads();

        // ---- acc += P x V over this wave's 8 D-tiles (32 WMMAs) ----
#pragma unroll
        for (int i = 0; i < 8; i++) {
            const unsigned short* vp =
                Vbase + (size_t)(d0 + i * 16 + mn) * S_DIM + j0;
#pragma unroll
            for (int f = 0; f < 4; f++) {
                BFrag vb;
                loadB(vb, vp + f * 32, g);
                acc[i] = __builtin_amdgcn_wmma_f32_16x16x32_bf16(
                    false, pA[f].v, false, vb.v, (short)0, acc[i], false, false);
            }
        }
    }

    // ---- epilogue: normalize and store f32 ----
#pragma unroll
    for (int i = 0; i < 8; i++) {
#pragma unroll
        for (int r = 0; r < 8; r++) {
            float o = acc[i][r] / lrow[r];
            size_t row = (size_t)bb * S_DIM + q0 + r + 8 * g;
            out[row * D_DIM + d0 + i * 16 + mn] = o;
        }
    }
}

// ---------------- host launch ----------------
extern "C" void kernel_launch(void* const* d_in, const int* in_sizes, int n_in,
                              void* d_out, int out_size, void* d_ws, size_t ws_size,
                              hipStream_t stream) {
    const float* x  = (const float*)d_in[0];
    const float* Wq = (const float*)d_in[1];
    const float* bq = (const float*)d_in[2];
    const float* Wk = (const float*)d_in[3];
    const float* bk = (const float*)d_in[4];
    const float* Wv = (const float*)d_in[5];
    const float* bv = (const float*)d_in[6];
    float* out = (float*)d_out;

    const int NX = B_DIM * S_DIM * D_DIM;   // 8388608
    const int NW = D_DIM * D_DIM;           // 1048576

    unsigned short* xb  = (unsigned short*)d_ws;
    unsigned short* Wqb = xb  + NX;
    unsigned short* Wkb = Wqb + NW;
    unsigned short* Wvb = Wkb + NW;
    unsigned short* Qb  = Wvb + NW;
    unsigned short* Kb  = Qb  + NX;
    unsigned short* Vt  = Kb  + NX;

    // Stage 1: conversions
    cvt_f32_bf16<<<(NX + 255) / 256, 256, 0, stream>>>(x, xb, NX);
    cvt_f32_bf16<<<(NW + 255) / 256, 256, 0, stream>>>(Wq, Wqb, NW);
    cvt_f32_bf16<<<(NW + 255) / 256, 256, 0, stream>>>(Wk, Wkb, NW);
    cvt_f32_bf16<<<(NW + 255) / 256, 256, 0, stream>>>(Wv, Wvb, NW);

    // Stage 2: QKV projection — 3 matrices * 2048 waves (64x64 block each)
    qkv_proj<<<768, 256, 0, stream>>>(xb, Wqb, Wkb, Wvb, bq, bk, bv, Qb, Kb, Vt);

    // Stage 3: flash attention — one block per 16-query tile
    flash_attn<<<B_DIM * (S_DIM / 16), 256, 0, stream>>>(Qb, Kb, Vt, out);
}
